// BaseWindowAttention_69973607187051
// MI455X (gfx1250) — matrix operand order repaired and checked by reading, not compiled, not run on hardware
//
#include <hip/hip_runtime.h>
#include <hip/hip_bf16.h>

typedef __bf16 bf16_t;
typedef __attribute__((ext_vector_type(16))) __bf16 v16bf;
typedef __attribute__((ext_vector_type(8)))  __bf16 v8bf;
typedef __attribute__((ext_vector_type(8)))  float   v8f;

#define DIM       256
#define HEADS     8
#define DIM_HEAD  32
#define L_        2
#define H_        256
#define W_        256
#define SCALE_F   0.17677669529663687f   /* 32^-0.5 */

#define XS 264   /* row stride (bf16 elems) of 64x256 LDS tiles (+8 pad) */
#define BS 68    /* bias row stride (f32) */
#define AS 72    /* row stride (bf16) for vT (256x64) */

__device__ __forceinline__ v16bf cat8(v8bf lo, v8bf hi) {
  v16bf r;
#pragma unroll
  for (int i = 0; i < 8; ++i) { r[i] = lo[i]; r[i + 8] = hi[i]; }
  return r;
}

// A-fragment (16x32 bf16, row-major source). Layout: lanes 0-15 hold row
// M=lane, K = 0..15; lanes 16-31 hold K = 16..31 (contiguous).
__device__ __forceinline__ v16bf load_a_frag(const bf16_t* base, int stride,
                                             int row0, int col0, int lane) {
  int m  = row0 + (lane & 15);
  int kb = (lane & 16) ? 16 : 0;
  const bf16_t* p = base + m * stride + col0 + kb;
  return cat8(*(const v8bf*)p, *(const v8bf*)(p + 8));
}

// B-fragment where this lane's 16 K-values are contiguous in memory.
__device__ __forceinline__ v16bf load_b_contig(const bf16_t* p) {
  return cat8(*(const v8bf*)p, *(const v8bf*)(p + 8));
}

__device__ __forceinline__ v8f wmma_bf16(v16bf a, v16bf b, v8f c) {
  return __builtin_amdgcn_wmma_f32_16x16x32_bf16(false, a, false, b, (short)0, c,
                                                 false, false);
}

// Attn scratch: 64x64 bf16 per wave, row stride 64, with an XOR swizzle on the
// 8-element column block (blk ^= (row>>1)&7). A-fragment b128 reads then hit 16
// distinct 4-dword bank groups across the 16 rows -> conflict-free.
__device__ __forceinline__ int attn_idx(int i, int j) {
  int blk = (j >> 3) ^ ((i >> 1) & 7);
  return i * 64 + (blk << 3) + (j & 7);
}

__device__ __forceinline__ v16bf load_a_attn(const bf16_t* base, int row0,
                                             int col0, int lane) {
  int m  = row0 + (lane & 15);
  int kb = (lane & 16) ? 16 : 0;
  int c0 = col0 + kb;                       // multiple of 8
  int sw = (m >> 1) & 7;
  const bf16_t* p0 = base + m * 64 + ((((c0      ) >> 3) ^ sw) << 3);
  const bf16_t* p1 = base + m * 64 + ((((c0 + 8) >> 3) ^ sw) << 3);
  return cat8(*(const v8bf*)p0, *(const v8bf*)p1);
}

// Pre-swizzle an fp32 weight matrix (K=256 rows, ncols columns) into bf16
// B-fragments: frag index = nt*8+kst; within a frag, lane L holds column
// nt*16+(L&15), K = kst*32 + ((L&16)?16:0) + e, e = 0..15 contiguous.
__global__ void pack_weights(const float* __restrict__ w, bf16_t* __restrict__ out,
                             int ncols, int ntiles_n) {
  int gid  = blockIdx.x * blockDim.x + threadIdx.x;
  int lane = gid & 31;
  int tile = gid >> 5;                     // tile = nt*8 + kst
  if (tile >= ntiles_n * 8) return;
  int nt = tile >> 3, kst = tile & 7;
  int kb = (lane & 16) ? 16 : 0;
  int n  = nt * 16 + (lane & 15);
  bf16_t* dst = out + (size_t)tile * 512 + lane * 16;
#pragma unroll
  for (int e = 0; e < 16; ++e) {
    int k = kst * 32 + kb + e;
    dst[e] = (bf16_t)w[(size_t)k * ncols + n];
  }
}

__global__ __launch_bounds__(256)
void win_attn_fused(const float* __restrict__ x,
                    const float* __restrict__ pos_emb,
                    const int*   __restrict__ rel_idx,
                    const bf16_t* __restrict__ wq_pk,
                    const bf16_t* __restrict__ wo_pk,
                    const float* __restrict__ b_out,
                    float* __restrict__ out) {
  // Single pool so the attn scratch can alias the dead sQ+sK region.
  // Layout (bytes):
  //   sX    : 64*XS*2            = 33792   (x tile, later concat-head output)
  //   sQ    : 64*XS*2            = 33792   \ aliased by per-wave attn scratch
  //   sK    : 64*XS*2            = 33792   / (8 waves * 8192 B = 65536 B)
  //   sVT   : 256*AS*2           = 36864   (v transposed: [c][token])
  //   sBias : 64*BS*4            = 17408
  // total = 155648 B  (<= 160 KB -> 2 workgroups per WGP)
  __shared__ __align__(16) char pool[155648];
  bf16_t* sX    = (bf16_t*)pool;
  bf16_t* sQ    = sX + 64 * XS;
  bf16_t* sK    = sQ + 64 * XS;
  bf16_t* sVT   = sK + 64 * XS;
  float*  sBias = (float*)(sVT + 256 * AS);
  bf16_t* sAttnPool = sQ;                  // reuse q/k region after barrier

  const int tid  = threadIdx.x;
  const int lane = tid & 31;
  const int wave = __builtin_amdgcn_readfirstlane(tid >> 5);

  const int blk   = blockIdx.x;           // 2048 = L * 32 * 32 windows
  const int layer = blk >> 10;
  const int widx  = blk & 1023;
  const int wy    = widx >> 5;
  const int wx    = widx & 31;

  const size_t win_base =
      (((size_t)layer * H_ + (size_t)wy * 8) * W_ + (size_t)wx * 8) * DIM;
  const float* xwin = x + win_base;

  // ---- Stage 0: load x window tile (64 tokens x 256) -> bf16 LDS; build bias
  for (int i = tid; i < 64 * 64; i += 256) {       // 4096 float4 loads
    int m  = i >> 6;                               // token 0..63
    int c4 = i & 63;                               // float4 index in row
    const float4 v =
        *(const float4*)(xwin + ((size_t)(m >> 3) * W_ + (m & 7)) * DIM + c4 * 4);
    bf16_t* d = &sX[m * XS + c4 * 4];
    d[0] = (bf16_t)v.x; d[1] = (bf16_t)v.y; d[2] = (bf16_t)v.z; d[3] = (bf16_t)v.w;
  }
  for (int i = tid; i < 64 * 64; i += 256) {
    int ii = i >> 6, jj = i & 63;
    int r0 = rel_idx[i * 2 + 0];
    int r1 = rel_idx[i * 2 + 1];
    sBias[ii * BS + jj] = pos_emb[r0 * 15 + r1];
  }
  __syncthreads();

  const int half8 = (lane & 16) ? 8 : 0;
  const int jlo   = lane & 15;

  // ---- Stage 1: QKV GEMM, out[64,768] = x[64,256] @ w_qkv
  //      q third -> sQ (scaled by SCALE), k third -> sK, v third -> sVT (transposed)
  for (int t = wave; t < 192; t += 8) {            // mt fixed per wave -> A frags hoist
    int mt = t & 3, nt = t >> 2;
    v8f acc = {};
#pragma unroll
    for (int kst = 0; kst < 8; ++kst) {
      v16bf a = load_a_frag(sX, XS, mt * 16, kst * 32, lane);
      v16bf b = load_b_contig(wq_pk + ((size_t)(nt * 8 + kst) << 9) + lane * 16);
      acc = wmma_bf16(a, b, acc);
    }
    int third = nt >> 4;
    int mrow  = mt * 16 + half8;
    if (third == 2) {
      int c = ((nt & 15) << 4) + jlo;              // head-dim column 0..255
      v8bf pk;
#pragma unroll
      for (int r = 0; r < 8; ++r) pk[r] = (bf16_t)acc[r];
      *(v8bf*)&sVT[c * AS + mrow] = pk;            // one 16B store per tile
    } else {
      bf16_t* dst = (third == 0) ? sQ : sK;
      float sc    = (third == 0) ? SCALE_F : 1.0f;
      int ncol = ((nt & 15) << 4) + jlo;
#pragma unroll
      for (int r = 0; r < 8; ++r)
        dst[(mrow + r) * XS + ncol] = (bf16_t)(acc[r] * sc);
    }
  }
  __syncthreads();

  // ---- Stage 2: attention, one head per wave
  const int h = wave;
  const int kb16 = (lane & 16) ? 16 : 0;
  v8f dots[4][4];
  v16bf aq[4];
#pragma unroll
  for (int mt = 0; mt < 4; ++mt)
    aq[mt] = load_a_frag(sQ, XS, mt * 16, h * DIM_HEAD, lane);   // K = 32
#pragma unroll
  for (int nt = 0; nt < 4; ++nt) {
    int j = nt * 16 + jlo;
    v16bf bk = load_b_contig(sK + j * XS + h * DIM_HEAD + kb16);
#pragma unroll
    for (int mt = 0; mt < 4; ++mt) {
      v8f cb;                                      // bias tile as accumulator init
#pragma unroll
      for (int r = 0; r < 8; ++r)
        cb[r] = sBias[(mt * 16 + half8 + r) * BS + nt * 16 + jlo];
      dots[mt][nt] = wmma_bf16(aq[mt], bk, cb);    // SCALE already folded into q
    }
  }
  __syncthreads();   // all waves done reading sQ/sK -> safe to overlay attn scratch

  // softmax over j (rows live per 16-lane half); store UNNORMALIZED exp,
  // keep 1/rowsum in registers (same lane/row mapping as the AV D-tile).
  bf16_t* sA = sAttnPool + wave * 64 * 64;         // 8 KB per wave, swizzled
  float rinv[4][8];
#pragma unroll
  for (int mt = 0; mt < 4; ++mt) {
#pragma unroll
    for (int r = 0; r < 8; ++r) {
      int i = mt * 16 + r + half8;
      float v0 = dots[mt][0][r];
      float v1 = dots[mt][1][r];
      float v2 = dots[mt][2][r];
      float v3 = dots[mt][3][r];
      float mx = fmaxf(fmaxf(v0, v1), fmaxf(v2, v3));
      mx = fmaxf(mx, __shfl_xor(mx, 1, 32));
      mx = fmaxf(mx, __shfl_xor(mx, 2, 32));
      mx = fmaxf(mx, __shfl_xor(mx, 4, 32));
      mx = fmaxf(mx, __shfl_xor(mx, 8, 32));
      const float l2e = 1.4426950408889634f;
      float p0 = __builtin_exp2f((v0 - mx) * l2e);
      float p1 = __builtin_exp2f((v1 - mx) * l2e);
      float p2 = __builtin_exp2f((v2 - mx) * l2e);
      float p3 = __builtin_exp2f((v3 - mx) * l2e);
      float s = p0 + p1 + p2 + p3;
      s += __shfl_xor(s, 1, 32);
      s += __shfl_xor(s, 2, 32);
      s += __shfl_xor(s, 4, 32);
      s += __shfl_xor(s, 8, 32);
      rinv[mt][r] = __builtin_amdgcn_rcpf(s);
      sA[attn_idx(i, 0  + jlo)] = (bf16_t)p0;
      sA[attn_idx(i, 16 + jlo)] = (bf16_t)p1;
      sA[attn_idx(i, 32 + jlo)] = (bf16_t)p2;
      sA[attn_idx(i, 48 + jlo)] = (bf16_t)p3;
    }
  }

  // AV: out_h[64,32] = exp[64,64] @ v[64,32], then scale rows by 1/sum
  v8f oacc[4][2];
#pragma unroll
  for (int mt = 0; mt < 4; ++mt)
#pragma unroll
    for (int nv = 0; nv < 2; ++nv) oacc[mt][nv] = (v8f){};
#pragma unroll
  for (int kst = 0; kst < 2; ++kst) {
    v16bf aa[4];
#pragma unroll
    for (int mt = 0; mt < 4; ++mt)
      aa[mt] = load_a_attn(sA, mt * 16, kst * 32, lane);
#pragma unroll
    for (int nv = 0; nv < 2; ++nv) {
      int c = h * DIM_HEAD + nv * 16 + jlo;        // vT row = head-dim column
      v16bf bv = load_b_contig(sVT + c * AS + kst * 32 + kb16);
#pragma unroll
      for (int mt = 0; mt < 4; ++mt)
        oacc[mt][nv] = wmma_bf16(aa[mt], bv, oacc[mt][nv]);
    }
  }
  // normalize + concat heads into sX (reuse)
#pragma unroll
  for (int mt = 0; mt < 4; ++mt)
#pragma unroll
    for (int nv = 0; nv < 2; ++nv) {
      int ncol = h * DIM_HEAD + nv * 16 + jlo;
#pragma unroll
      for (int r = 0; r < 8; ++r)
        sX[(mt * 16 + half8 + r) * XS + ncol] = (bf16_t)(oacc[mt][nv][r] * rinv[mt][r]);
    }
  __syncthreads();

  // ---- Stage 3: projection, out[64,256] = o[64,256] @ w_out + b_out (fp32 store)
  float* owin = out + win_base;
  for (int t = wave; t < 64; t += 8) {             // mt fixed per wave -> A frags hoist
    int mt = t & 3, nt = t >> 2;
    v8f acc = {};
#pragma unroll
    for (int kst = 0; kst < 8; ++kst) {
      v16bf a = load_a_frag(sX, XS, mt * 16, kst * 32, lane);
      v16bf b = load_b_contig(wo_pk + ((size_t)(nt * 8 + kst) << 9) + lane * 16);
      acc = wmma_bf16(a, b, acc);
    }
    int n  = nt * 16 + jlo;
    float bo = b_out[n];
    int mrow = mt * 16 + half8;
#pragma unroll
    for (int r = 0; r < 8; ++r) {
      int m = mrow + r;
      owin[((size_t)(m >> 3) * W_ + (m & 7)) * DIM + n] = acc[r] + bo;
    }
  }
}

extern "C" void kernel_launch(void* const* d_in, const int* in_sizes, int n_in,
                              void* d_out, int out_size, void* d_ws, size_t ws_size,
                              hipStream_t stream) {
  const float* x       = (const float*)d_in[0];
  const float* w_qkv   = (const float*)d_in[1];
  const float* pos_emb = (const float*)d_in[2];
  const float* w_out   = (const float*)d_in[3];
  const float* b_out   = (const float*)d_in[4];
  const int*   rel_idx = (const int*)d_in[5];
  float* outp = (float*)d_out;

  bf16_t* wq_pk = (bf16_t*)d_ws;                       // 48*8*512 bf16 = 384 KiB
  bf16_t* wo_pk = wq_pk + (size_t)48 * 8 * 512;        // 16*8*512 bf16 = 128 KiB

  pack_weights<<<(48 * 8 * 32 + 255) / 256, 256, 0, stream>>>(w_qkv, wq_pk, 768, 48);
  pack_weights<<<(16 * 8 * 32 + 255) / 256, 256, 0, stream>>>(w_out, wo_pk, 256, 16);

  win_attn_fused<<<2048, 256, 0, stream>>>(x, pos_emb, rel_idx, wq_pk, wo_pk,
                                           b_out, outp);
}
